// Model_73409581023751
// MI455X (gfx1250) — matrix operand order repaired and checked by reading, not compiled, not run on hardware
//
#include <hip/hip_runtime.h>
#include <cstdint>
#include <cstddef>

// ---------------- problem constants ----------------
#define GZ     15
#define CELLS  (GZ*GZ*GZ)        // 3375
#define BATCH  4
#define NPTS   262144            // 2^18
#define S1     512               // 8^3 conv spatial per sample
#define NF     (BATCH*S1)        // 2048  (forward GEMM N)
#define NSEED  BATCH
#define NB     (NSEED*NF)        // 8192  (backward GEMM N, 4 seeds batched)
#define K0     512               // conv1 K (4*125=500 padded)
#define K1     1024              // conv2..7 K (8*125=1000 padded)

typedef __attribute__((ext_vector_type(16))) _Float16 v16h;
typedef __attribute__((ext_vector_type(8)))  float    v8f;
typedef __attribute__((ext_vector_type(4))) unsigned int u32x4;
typedef __attribute__((ext_vector_type(8)))  int      i32x8;
typedef __attribute__((ext_vector_type(4)))  int      i32x4;

#if defined(__HIP_DEVICE_COMPILE__) && __has_builtin(__builtin_amdgcn_tensor_load_to_lds) && __has_builtin(__builtin_amdgcn_s_wait_tensorcnt)
#define USE_TDM 1
#else
#define USE_TDM 0
#endif

// ---------------- small device helpers ----------------
__device__ __forceinline__ float eluf(float x){ return x > 0.f ? x : (__expf(x) - 1.f); }
__device__ __forceinline__ float elup(float x){ return x > 0.f ? 1.f : __expf(x); }

// order-preserving float->u32 key (for packed argmax via atomicMax on u64)
__device__ __forceinline__ unsigned fkey(float f){
    unsigned u = __float_as_uint(f);
    return (u & 0x80000000u) ? ~u : (u | 0x80000000u);
}
__device__ __forceinline__ float funkey(unsigned k){
    unsigned u = (k & 0x80000000u) ? (k & 0x7FFFFFFFu) : ~k;
    return __uint_as_float(u);
}

// MLP weights packed in shared: w1[8] b1[4] w2[16] b2[4] w3[8] b3[2] = 42 floats
__device__ __forceinline__ void load_mlpw(float* sw, int tid,
    const float* w1, const float* b1, const float* w2,
    const float* b2, const float* w3, const float* b3)
{
    if      (tid <  8) sw[tid] = w1[tid];
    else if (tid < 12) sw[tid] = b1[tid-8];
    else if (tid < 28) sw[tid] = w2[tid-12];
    else if (tid < 32) sw[tid] = b2[tid-28];
    else if (tid < 40) sw[tid] = w3[tid-32];
    else if (tid < 42) sw[tid] = b3[tid-40];
}

// ---------------- init: packed cells + accumulators ----------------
__global__ void k_init(unsigned long long* cells, float* accum){
    int idx = blockIdx.x*256 + threadIdx.x;
    if (idx < BATCH*4*CELLS) cells[idx] = 0x80000000FFFFFFFFull; // key(0.0f) | invalid idx
    if (idx < 8) accum[idx] = 0.f;
}

// ---------------- point stream: disty -> MLP -> packed scatter-argmax ----------------
__global__ void k_points(const float* __restrict__ pts,
                         const float* __restrict__ w1, const float* __restrict__ b1,
                         const float* __restrict__ w2, const float* __restrict__ b2,
                         const float* __restrict__ w3, const float* __restrict__ b3,
                         unsigned long long* __restrict__ cells,
                         int chbase, float* accum_sumsq)
{
    __shared__ float sw[42];
    __shared__ float red[256];
    int tid = threadIdx.x;
    load_mlpw(sw, tid, w1, b1, w2, b2, w3, b3);
    __syncthreads();

    long idx = (long)blockIdx.x*256 + tid;        // grid covers exactly BATCH*NPTS
    int b = (int)(idx >> 18);
    unsigned n = (unsigned)(idx & (NPTS-1));
    const float* p = pts + idx*3;
    __builtin_prefetch(p + 3*4096, 0, 1);         // -> global_prefetch_b8

    float px = p[0], py = p[1], pz = p[2];
    float ox = px - (floorf(px)+0.5f);
    float oy = py - (floorf(py)+0.5f);
    float oz = pz - (floorf(pz)+0.5f);
    float dn = sqrtf(ox*ox+oy*oy+oz*oz + 1e-12f);
    float xm = (ox+oy+oz)*(1.f/3.f);

    float a1[4], a2[4];
    #pragma unroll
    for (int j=0;j<4;++j) a1[j] = eluf(dn*sw[j] + xm*sw[4+j] + sw[8+j]);
    #pragma unroll
    for (int j=0;j<4;++j){
        float t = sw[28+j];
        #pragma unroll
        for (int i=0;i<4;++i) t += a1[i]*sw[12+i*4+j];
        a2[j] = eluf(t);
    }
    float y0 = sw[40], y1 = sw[41];
    #pragma unroll
    for (int k=0;k<4;++k){ y0 += a2[k]*sw[32+2*k]; y1 += a2[k]*sw[32+2*k+1]; }

    int vx = (int)floorf(px); vx = vx<0?0:(vx>14?14:vx);
    int vy = (int)floorf(py); vy = vy<0?0:(vy>14?14:vy);
    int vz = (int)floorf(pz); vz = vz<0?0:(vz>14?14:vz);
    int flat = (vx*GZ + vy)*GZ + vz;

    unsigned long long pk0 = ((unsigned long long)fkey(y0) << 32) | n;
    unsigned long long pk1 = ((unsigned long long)fkey(y1) << 32) | n;
    atomicMax(&cells[((size_t)(b*4 + chbase + 0))*CELLS + flat], pk0);
    atomicMax(&cells[((size_t)(b*4 + chbase + 1))*CELLS + flat], pk1);

    if (accum_sumsq){  // fused sum of goals^2 for the cost term
        red[tid] = px*px + py*py + pz*pz;
        __syncthreads();
        for (int s=128; s>0; s>>=1){ if (tid<s) red[tid]+=red[tid+s]; __syncthreads(); }
        if (tid==0) atomicAdd(accum_sumsq, red[0]);
    }
}

// ---------------- unpack scene grid values ----------------
__global__ void k_scene(const unsigned long long* __restrict__ cells, float* __restrict__ scene){
    int idx = blockIdx.x*256 + threadIdx.x;
    if (idx >= BATCH*4*CELLS) return;
    scene[idx] = funkey((unsigned)(cells[idx] >> 32));
}

// ---------------- pack weights into WMMA A-fragment layout ----------------
// A-frag (16x32 f16): lane l, half j -> m=l&15, k = 32*step + ((l>>4)<<4) + j
// fwd:  A[m][k] = w[m][k/125][k%125]                (m=oc)
// bwd:  A[m][k] = w[k/125][m][124 - k%125]          (m=ic, flipped kernel, transposed)
__global__ void k_packA(const float* __restrict__ w, _Float16* __restrict__ Ap,
                        int cin, int Ksteps, int flip)
{
    int idx = blockIdx.x*256 + threadIdx.x;
    if (idx >= Ksteps*512) return;
    int s = idx >> 9, rem = idx & 511, lane = rem >> 4, j = rem & 15;
    int m = lane & 15;
    int k = s*32 + ((lane>>4)<<4) + j;
    int Kreal = cin*125;
    float v = 0.f;
    if (m < 8 && k < Kreal){
        int a = k/125, r = k%125;
        v = flip ? w[((size_t)a*cin + m)*125 + (124 - r)]
                 : w[((size_t)m*cin + a)*125 + r];
    }
    Ap[idx] = (_Float16)v;
}

// ---------------- im2col: conv1 (scene 4ch 15^3, stride2 pad2, K=512) ----------------
__global__ void k_im2col0(const float* __restrict__ scene, _Float16* __restrict__ Bm){
    int idx = blockIdx.x*256 + threadIdx.x;        // NF*K0 threads
    int n = idx >> 9, k = idx & 511;
    int b = n >> 9, pos = n & 511;
    int oz = pos>>6, oy = (pos>>3)&7, ox = pos&7;
    float v = 0.f;
    if (k < 500){
        int ic = k/125, r = k%125, kz = r/25, ky = (r/5)%5, kx = r%5;
        int z = oz*2-2+kz, y = oy*2-2+ky, x = ox*2-2+kx;
        if ((unsigned)z < 15u && (unsigned)y < 15u && (unsigned)x < 15u)
            v = scene[((size_t)(b*4+ic))*CELLS + (z*GZ+y)*GZ + x];
    }
    Bm[(size_t)n*K0 + k] = (_Float16)v;
}

// ---------------- im2col: 8ch 8^3 pad2 (fwd mid layers and bwd transpose-conv) ----------------
// src layout [c][Nrows]; rows n keep (seed,b) base intact
__global__ void k_im2col(const float* __restrict__ src, _Float16* __restrict__ Bm, int Nrows){
    size_t idx = (size_t)blockIdx.x*256 + threadIdx.x;   // Nrows*K1 threads
    int n = (int)(idx >> 10), k = (int)(idx & 1023);
    int pos = n & 511, base = n - pos;
    int oz = pos>>6, oy = (pos>>3)&7, ox = pos&7;
    float v = 0.f;
    if (k < 1000){
        int ic = k/125, r = k%125, kz = r/25, ky = (r/5)%5, kx = r%5;
        int z = oz-2+kz, y = oy-2+ky, x = ox-2+kx;
        if ((unsigned)z < 8u && (unsigned)y < 8u && (unsigned)x < 8u)
            v = src[(size_t)ic*Nrows + base + (z*8+y)*8 + x];
    }
    Bm[(size_t)n*K1 + k] = (_Float16)v;
}

// ---------------- WMMA GEMM: D[16 x Ntot] = Apack(16xK) * Bmat(KxNtot) (+bias) ----------------
// 8 waves / block, one 16x16 N-tile per wave; A staged in LDS via Tensor Data Mover.
__global__ void __launch_bounds__(256) k_gemm(const _Float16* __restrict__ Apack,
                                              const _Float16* __restrict__ Bmat,
                                              const float* __restrict__ bias,
                                              float* __restrict__ Dout,
                                              int Ksteps, int Ntot)
{
    __shared__ __align__(32) _Float16 As[16384];  // up to 32KB of packed A
    int nelem8 = Ksteps*128;                      // 8-byte elements
#if USE_TDM
    if (threadIdx.x == 0){
        unsigned lds = (unsigned)(size_t)&As[0];                // low 32 bits = LDS offset
        unsigned long long ga = (unsigned long long)(size_t)Apack;
        u32x4 g0; // D# group0: count=1 | lds_addr | global_addr | type=2
        g0[0] = 1u;
        g0[1] = lds;
        g0[2] = (unsigned)ga;
        g0[3] = (unsigned)((ga>>32) & 0x1FFFFFFull) | (2u<<30);
        i32x8 g1; // group1: data_size=8B, 1-D tensor dim0=tile_dim0=nelem8, stride0=nelem8
        g1[0] = 0x00030000;
        g1[1] = (int)(((unsigned)nelem8 & 0xFFFFu) << 16);
        g1[2] = (int)(((unsigned)nelem8 >> 16) | (1u<<16));
        g1[3] = (int)((unsigned)nelem8 << 16);
        g1[4] = 1; g1[5] = nelem8; g1[6] = 0; g1[7] = 0;
        i32x4 gz4; gz4[0]=0; gz4[1]=0; gz4[2]=0; gz4[3]=0;
        i32x8 gz8; gz8[0]=0; gz8[1]=0; gz8[2]=0; gz8[3]=0;
        gz8[4]=0; gz8[5]=0; gz8[6]=0; gz8[7]=0;
        // clang-23 / therock-10.0 6-arg form: (g0, g1, g2, g3, g4, cpol)
        __builtin_amdgcn_tensor_load_to_lds(g0, g1, gz4, gz4, gz8, 0); // tensor_load_to_lds
        __builtin_amdgcn_s_wait_tensorcnt(0);                          // s_wait_tensorcnt 0
    }
#else
    for (int i = threadIdx.x; i < nelem8; i += 256)
        ((unsigned long long*)As)[i] = ((const unsigned long long*)Apack)[i];
#endif
    __syncthreads();

    int lane = threadIdx.x & 31, wid = threadIdx.x >> 5;
    int ntile = blockIdx.x*8 + wid;
    int ncol  = ntile*16 + (lane & 15);
    const _Float16* brow = Bmat + (size_t)ncol*(Ksteps*32) + (lane>>4)*16;

    v8f acc = {};
    for (int s = 0; s < Ksteps; ++s){
        v16h a  = *(const v16h*)(As + s*512 + lane*16);   // 32B contiguous per lane
        v16h bf = *(const v16h*)(brow + s*32);            // 32B contiguous per lane
        acc = __builtin_amdgcn_wmma_f32_16x16x32_f16(false, a, false, bf,
                                                     (short)0, acc, false, false);
    }
    if (lane < 16){   // lanes 0-15 hold M=0..7 (real output channels)
        bool hb = (bias != nullptr);
        #pragma unroll
        for (int r = 0; r < 8; ++r){
            float v = acc[r];
            if (hb) v += bias[r];
            Dout[(size_t)r*Ntot + ncol] = v;
        }
    }
}

// ---------------- BN forward (training mode, biased var) + ELU ----------------
__global__ void k_bn_fwd(const float* __restrict__ z, const float* __restrict__ g,
                         const float* __restrict__ beta, float* __restrict__ xhat,
                         float* __restrict__ act, float* __restrict__ rstd_out)
{
    int c = blockIdx.x, tid = threadIdx.x;
    const float* zc = z + (size_t)c*NF;
    float s = 0.f, ss = 0.f;
    for (int i = tid; i < NF; i += 256){ float v = zc[i]; s += v; ss += v*v; }
    __shared__ float rs[256], rq[256];
    __shared__ float smu, srst;
    rs[tid] = s; rq[tid] = ss; __syncthreads();
    for (int o=128;o>0;o>>=1){ if (tid<o){ rs[tid]+=rs[tid+o]; rq[tid]+=rq[tid+o]; } __syncthreads(); }
    if (tid==0){
        float mu = rs[0]*(1.f/NF);
        float var = rq[0]*(1.f/NF) - mu*mu;
        float r = rsqrtf(var + 1e-5f);
        smu = mu; srst = r; rstd_out[c] = r;
    }
    __syncthreads();
    float gg = g[c], bb = beta[c];
    for (int i = tid; i < NF; i += 256){
        float xh = (zc[i]-smu)*srst;
        xhat[(size_t)c*NF+i] = xh;
        float bo = xh*gg + bb;
        act[(size_t)c*NF+i] = bo > 0.f ? bo : (__expf(bo)-1.f);
    }
}

// ---------------- ELU' + BN backward (per seed s, per channel c) ----------------
__global__ void k_bn_bwd(const float* __restrict__ dA, const float* __restrict__ xhat,
                         const float* __restrict__ g, const float* __restrict__ beta,
                         const float* __restrict__ rstd, float* __restrict__ dz)
{
    int q = blockIdx.x; int s = q>>3, c = q&7; int tid = threadIdx.x;
    size_t base = (size_t)c*NB + (size_t)s*NF;
    const float* xh = xhat + (size_t)c*NF;
    float gg = g[c], bb = beta[c], rr = rstd[c];
    float s1 = 0.f, s2 = 0.f;
    for (int i = tid; i < NF; i += 256){
        float x  = xh[i];
        float bo = x*gg + bb;
        float dxh = dA[base+i] * (bo > 0.f ? 1.f : __expf(bo)) * gg;
        dz[base+i] = dxh;
        s1 += dxh; s2 += dxh*x;
    }
    __shared__ float r1[256], r2[256];
    r1[tid]=s1; r2[tid]=s2; __syncthreads();
    for (int o=128;o>0;o>>=1){ if (tid<o){ r1[tid]+=r1[tid+o]; r2[tid]+=r2[tid+o]; } __syncthreads(); }
    float m1 = r1[0]*(1.f/NF), m2 = r2[0]*(1.f/NF);
    for (int i = tid; i < NF; i += 256)
        dz[base+i] = rr*(dz[base+i] - m1 - xh[i]*m2);
}

// ---------------- seed dAct of last layer: d mean_b / d act7 ----------------
__global__ void k_seed(float* __restrict__ dA){
    int idx = blockIdx.x*256 + threadIdx.x;      // 8*NB
    int np = idx & (NB-1);
    int s = np >> 11, b = (np >> 9) & 3;
    dA[idx] = (s == b) ? (1.0f/4096.0f) : 0.0f;
}

// ---------------- conv1 transpose (stride 2) back to scene grad, channels 0/1, b==seed ----------------
__global__ void k_conv1_bwd(const float* __restrict__ dz, const float* __restrict__ w0,
                            float* __restrict__ dscene)
{
    int tid = blockIdx.x*256 + threadIdx.x;
    if (tid >= NSEED*2*CELLS) return;
    int s = tid/(2*CELLS); int rem = tid%(2*CELLS);
    int c = rem/CELLS; int cell = rem%CELLS;
    int zc = cell/(GZ*GZ), yc = (cell/GZ)%GZ, xc = cell%GZ;
    float acc = 0.f;
    for (int oz=0; oz<8; ++oz){ int kz = zc+2-2*oz; if ((unsigned)kz > 4u) continue;
     for (int oy=0; oy<8; ++oy){ int ky = yc+2-2*oy; if ((unsigned)ky > 4u) continue;
      for (int ox=0; ox<8; ++ox){ int kx = xc+2-2*ox; if ((unsigned)kx > 4u) continue;
        int opos = (oz*8+oy)*8+ox;
        int kk = kz*25+ky*5+kx;
        #pragma unroll
        for (int oc=0; oc<8; ++oc)
            acc += w0[((size_t)oc*4 + c)*125 + kk] *
                   dz[(size_t)oc*NB + (size_t)s*NF + (size_t)s*S1 + opos];
      }}}
    dscene[((size_t)s*2 + c)*CELLS + cell] = acc;
}

// ---------------- scatter-max backward: winner points -> MLP backprop -> reductions ----------------
__device__ void backprop_point(const float* sw, const float* pts, const float* goals,
                               int b, unsigned n, float dy0, float dy1, float* accum)
{
    const float* p = pts + ((size_t)b*NPTS + n)*3;
    float off[3]; float z1[4], a1[4], z2[4];
    float ssum = 0.f;
    #pragma unroll
    for (int i=0;i<3;++i){ float o = p[i] - (floorf(p[i])+0.5f); off[i]=o; ssum += o*o; }
    float dn = sqrtf(ssum + 1e-12f);
    float xm = (off[0]+off[1]+off[2])*(1.f/3.f);
    #pragma unroll
    for (int j=0;j<4;++j){ z1[j] = dn*sw[j] + xm*sw[4+j] + sw[8+j]; a1[j] = eluf(z1[j]); }
    #pragma unroll
    for (int j=0;j<4;++j){
        float t = sw[28+j];
        #pragma unroll
        for (int i=0;i<4;++i) t += a1[i]*sw[12+i*4+j];
        z2[j] = t;
    }
    float dz2[4], dz1[4];
    #pragma unroll
    for (int k=0;k<4;++k){
        float da2 = dy0*sw[32+2*k] + dy1*sw[32+2*k+1];
        dz2[k] = da2*elup(z2[k]);
    }
    #pragma unroll
    for (int i=0;i<4;++i){
        float da1 = 0.f;
        #pragma unroll
        for (int k=0;k<4;++k) da1 += dz2[k]*sw[12+i*4+k];
        dz1[i] = da1*elup(z1[i]);
    }
    float dd = 0.f, dm = 0.f;
    #pragma unroll
    for (int j=0;j<4;++j){ dd += dz1[j]*sw[j]; dm += dz1[j]*sw[4+j]; }
    const float* gp = goals + ((size_t)b*NPTS + n)*3;
    float cs = 0.f, gs = 0.f, ga = 0.f;
    #pragma unroll
    for (int i=0;i<3;++i){
        float dv = dd*off[i]/dn + dm*(1.f/3.f);
        gs += dv; ga += fabsf(dv);
        cs += dv*dv - 2.f*dv*gp[i];
    }
    atomicAdd(accum+1, cs);
    atomicAdd(accum+2, gs);
    atomicAdd(accum+3, ga);
}

__global__ void k_scatter(const unsigned long long* __restrict__ cells,
                          const float* __restrict__ dscene,
                          const float* __restrict__ pts, const float* __restrict__ goals,
                          const float* w1, const float* b1, const float* w2,
                          const float* b2, const float* w3, const float* b3,
                          float* accum)
{
    __shared__ float sw[42];
    int tid = threadIdx.x;
    load_mlpw(sw, tid, w1, b1, w2, b2, w3, b3);
    __syncthreads();
    int idx = blockIdx.x*256 + tid;
    if (idx >= BATCH*CELLS) return;
    int b = idx / CELLS, cell = idx % CELLS;
    unsigned long long e0 = cells[((size_t)(b*4)+0)*CELLS + cell];
    unsigned long long e1 = cells[((size_t)(b*4)+1)*CELLS + cell];
    unsigned i0 = (unsigned)e0, i1 = (unsigned)e1;
    bool v0 = (i0 != 0xFFFFFFFFu), v1 = (i1 != 0xFFFFFFFFu);
    float gy0 = dscene[((size_t)b*2+0)*CELLS + cell];
    float gy1 = dscene[((size_t)b*2+1)*CELLS + cell];
    if (v0 && v1 && i0 == i1){
        backprop_point(sw, pts, goals, b, i0, gy0, gy1, accum);
    } else {
        if (v0) backprop_point(sw, pts, goals, b, i0, gy0, 0.f, accum);
        if (v1) backprop_point(sw, pts, goals, b, i1, 0.f, gy1, accum);
    }
}

__global__ void k_fin(const float* __restrict__ accum, float* __restrict__ out){
    out[0] = (accum[0] + accum[1]) * (1.0f/(float)(3ll*NPTS*BATCH));
    out[1] = accum[2]*0.25f;
    out[2] = accum[3]*0.25f;
}

// ---------------- host orchestration ----------------
extern "C" void kernel_launch(void* const* d_in, const int* in_sizes, int n_in,
                              void* d_out, int out_size, void* d_ws, size_t ws_size,
                              hipStream_t stream)
{
    (void)in_sizes; (void)n_in; (void)out_size; (void)ws_size;
    const float* ipts = (const float*)d_in[0];
    const float* gpts = (const float*)d_in[1];
    const float* w1 = (const float*)d_in[2]; const float* b1 = (const float*)d_in[3];
    const float* w2 = (const float*)d_in[4]; const float* b2 = (const float*)d_in[5];
    const float* w3 = (const float*)d_in[6]; const float* b3 = (const float*)d_in[7];
    const float *cw[7], *cb[7], *cg[7], *cbe[7];
    for (int i = 0; i < 7; ++i){
        cw[i]  = (const float*)d_in[8 + 4*i + 0];
        cb[i]  = (const float*)d_in[8 + 4*i + 1];
        cg[i]  = (const float*)d_in[8 + 4*i + 2];
        cbe[i] = (const float*)d_in[8 + 4*i + 3];
    }

    char* base = (char*)d_ws;
    size_t off = 0;
    auto alloc = [&](size_t bytes)->void*{
        void* p = base + off;
        off = (off + bytes + 255) & ~(size_t)255;
        return p;
    };
    float*              accum  = (float*)alloc(64);
    unsigned long long* cells  = (unsigned long long*)alloc((size_t)BATCH*4*CELLS*8);
    float*              scene  = (float*)alloc((size_t)BATCH*4*CELLS*4);
    _Float16* ApF[7];
    ApF[0] = (_Float16*)alloc((size_t)16*512*2);
    for (int i = 1; i < 7; ++i) ApF[i] = (_Float16*)alloc((size_t)32*512*2);
    _Float16* ApB[6];
    for (int i = 0; i < 6; ++i) ApB[i] = (_Float16*)alloc((size_t)32*512*2);
    _Float16* Bmat  = (_Float16*)alloc((size_t)NB*K1*2);     // 16 MB, shared fwd/bwd
    float*    zout  = (float*)alloc((size_t)8*NF*4);
    float*    xhat  = (float*)alloc((size_t)7*8*NF*4);
    float*    act   = (float*)alloc((size_t)8*NF*4);
    float*    rstd  = (float*)alloc((size_t)7*8*4);
    float*    dA    = (float*)alloc((size_t)8*NB*4);
    float*    dBuf  = (float*)alloc((size_t)8*NB*4);
    float*    dzb   = (float*)alloc((size_t)8*NB*4);
    float*    dscene= (float*)alloc((size_t)NSEED*2*CELLS*4);

    // 1) init cells + accumulators
    k_init<<<(BATCH*4*CELLS + 255)/256, 256, 0, stream>>>(cells, accum);
    // 2) point streams (inputs -> ch0/1, goals -> ch2/3 fused with sum(goals^2))
    k_points<<<(BATCH*NPTS)/256, 256, 0, stream>>>(ipts, w1,b1,w2,b2,w3,b3, cells, 0, nullptr);
    k_points<<<(BATCH*NPTS)/256, 256, 0, stream>>>(gpts, w1,b1,w2,b2,w3,b3, cells, 2, accum);
    // 3) unpack scene values
    k_scene<<<(BATCH*4*CELLS + 255)/256, 256, 0, stream>>>(cells, scene);
    // 4) pack conv weights into WMMA A-fragments (fwd and flipped/transposed bwd)
    for (int i = 0; i < 7; ++i){
        int cin = i ? 8 : 4, Ks = i ? 32 : 16;
        k_packA<<<(Ks*512 + 255)/256, 256, 0, stream>>>(cw[i], ApF[i], cin, Ks, 0);
    }
    for (int i = 1; i < 7; ++i)
        k_packA<<<64, 256, 0, stream>>>(cw[i], ApB[i-1], 8, 32, 1);
    // 5) forward conv stack (im2col -> WMMA GEMM -> BN+ELU)
    for (int i = 0; i < 7; ++i){
        if (i == 0) k_im2col0<<<(NF*K0)/256, 256, 0, stream>>>(scene, Bmat);
        else        k_im2col<<<(NF*K1)/256, 256, 0, stream>>>(act, Bmat, NF);
        k_gemm<<<NF/128, 256, 0, stream>>>(ApF[i], Bmat, cb[i], zout, i ? 32 : 16, NF);
        k_bn_fwd<<<8, 256, 0, stream>>>(zout, cg[i], cbe[i], xhat + (size_t)i*8*NF, act, rstd + i*8);
    }
    // 6) backward: 4 seeds batched into N=8192 WMMA GEMM chain
    k_seed<<<(8*NB)/256, 256, 0, stream>>>(dA);
    float* cur = dA; float* nxt = dBuf;
    for (int i = 6; i >= 1; --i){
        k_bn_bwd<<<32, 256, 0, stream>>>(cur, xhat + (size_t)i*8*NF, cg[i], cbe[i], rstd + i*8, dzb);
        k_im2col<<<(int)(((size_t)NB*K1)/256), 256, 0, stream>>>(dzb, Bmat, NB);
        k_gemm<<<NB/128, 256, 0, stream>>>(ApB[i-1], Bmat, nullptr, nxt, 32, NB);
        float* t = cur; cur = nxt; nxt = t;
    }
    k_bn_bwd<<<32, 256, 0, stream>>>(cur, xhat, cg[0], cbe[0], rstd, dzb);
    k_conv1_bwd<<<(NSEED*2*CELLS + 255)/256, 256, 0, stream>>>(dzb, cw[0], dscene);
    // 7) winner backprop + reductions, then finalize the 3 scalars
    k_scatter<<<(BATCH*CELLS + 255)/256, 256, 0, stream>>>(cells, dscene, ipts, gpts,
                                                           w1,b1,w2,b2,w3,b3, accum);
    k_fin<<<1, 1, 0, stream>>>(accum, (float*)d_out);
}
